// Attention_82377472738010
// MI455X (gfx1250) — matrix operand order repaired and checked by reading, not compiled, not run on hardware
//
#include <hip/hip_runtime.h>
#include <hip/hip_bf16.h>

typedef _Float16 half_t;
typedef __attribute__((ext_vector_type(16))) _Float16 v16h;
typedef __attribute__((ext_vector_type(8)))  _Float16 v8h;
typedef __attribute__((ext_vector_type(8)))  float    v8f;

#define B_   64
#define S_   512
#define D_   100
#define H_   4
#define DH_  8
#define SZ_  32
#define DFF_ 400
#define M_   (B_ * S_)   // 32768 rows
#define DP_  128         // D padded to multiple of 32 (WMMA K)
#define FFP_ 416         // DFF padded to multiple of 32

// ---------------------------------------------------------------------------
// WMMA helpers (CDNA5 f16 -> f32, 16x16x32)
// ---------------------------------------------------------------------------
__device__ __forceinline__ v8f wmma16(v16h a, v16h b, v8f c) {
    return __builtin_amdgcn_wmma_f32_16x16x32_f16(
        /*neg_a=*/false, a, /*neg_b=*/false, b,
        /*c_mod=*/(short)0, c, /*reuse_a=*/false, /*reuse_b=*/false);
}

// Full A fragment (16x32 f16): lane L holds row (L&15); elems j = K 16h+j,
// h = L>>4.  Two 16B contiguous loads.
__device__ __forceinline__ v16h load_a(const half_t* base, int pitch,
                                       int row0, int kblk, int lane) {
    int r  = row0 + (lane & 15);
    int hh = lane >> 4;
    const half_t* p = base + (size_t)r * pitch + kblk + 16 * hh;
    v8h lo = *(const v8h*)p;
    v8h hi = *(const v8h*)(p + 8);
    v16h out;
#pragma unroll
    for (int i = 0; i < 8; ++i) { out[i] = lo[i]; out[8 + i] = hi[i]; }
    return out;
}

// Full B fragment (32x16 f16): lane L = column N (L&15); elems j = K 16h+j.
// base is row-major [N, Kpitch]; one 32B contiguous load.
__device__ __forceinline__ v16h load_b(const half_t* base, int pitch,
                                       int n0, int kblk, int lane) {
    int n  = n0 + (lane & 15);
    int hh = lane >> 4;
    return *(const v16h*)(base + (size_t)n * pitch + kblk + 16 * hh);
}

// K-dim==8 fragment from an unpadded pitch-8 matrix [rows, 8].
// Works for BOTH A (row = idx) and B (col = idx): only lanes 0..15 have K<8
// data (A: elems 0..7 = K 0..7; B: elems 0..7 = K 0..7); rest must be zero.
__device__ __forceinline__ v16h load_frag_k8(const half_t* base, int idx0, int lane) {
    v16h out = {};
    if ((lane >> 4) == 0) {
        v8h lo = *(const v8h*)(base + (size_t)(idx0 + lane) * DH_);
#pragma unroll
        for (int i = 0; i < 8; ++i) out[i] = lo[i];
    }
    return out;
}

// B fragment from VT [8, S_]: col n = dim (valid n<8), elems j = key 16h+j.
__device__ __forceinline__ v16h load_b_vt(const half_t* vt, int kblk, int lane) {
    v16h out = {};
    int n = lane & 15, hh = lane >> 4;
    if (n < DH_)
        out = *(const v16h*)(vt + (size_t)n * S_ + kblk + 16 * hh);
    return out;
}

// ---------------------------------------------------------------------------
// 0) fp32 -> f16 into a padded rectangle; pads written as zeros (no memset)
// ---------------------------------------------------------------------------
__global__ void cvt_pad(const float* __restrict__ src, half_t* __restrict__ dst,
                        int rows, int cols, int prows, int pitch) {
    int idx = blockIdx.x * blockDim.x + threadIdx.x;
    if (idx < prows * pitch) {
        int r = idx / pitch, c = idx - r * pitch;
        float v = (r < rows && c < cols) ? src[r * cols + c] : 0.f;
        dst[idx] = (half_t)v;
    }
}

// ---------------------------------------------------------------------------
// 1) QKV projections: x16[M,128] @ W^T.  Q,K -> [B,H,S,8]; V -> [B,H,8,S].
// ---------------------------------------------------------------------------
__global__ __launch_bounds__(32) void qkv_kernel(
    const half_t* __restrict__ x16,
    const half_t* __restrict__ Wq16, const half_t* __restrict__ Wk16,
    const half_t* __restrict__ Wv16,
    const float* __restrict__ bq, const float* __restrict__ bk,
    const float* __restrict__ bv,
    half_t* __restrict__ Q16, half_t* __restrict__ K16, half_t* __restrict__ VT16) {
    int lane = threadIdx.x;
    int row0 = blockIdx.x * 16;
    int hh = lane >> 4, ln = lane & 15;
    int bb = row0 >> 9;              // batch (tile never crosses batch: 16|512)
    int s0 = row0 & (S_ - 1);
    v16h a[4];
#pragma unroll
    for (int kb = 0; kb < 4; ++kb) a[kb] = load_a(x16, DP_, row0, kb * 32, lane);

#pragma unroll
    for (int nt = 0; nt < 2; ++nt) {
        int n = nt * 16 + ln;            // feature 0..31
        int hd = n >> 3, dh = n & 7;     // head, dim-in-head
        size_t qkbase = ((size_t)(bb * H_ + hd) * S_ + s0) * DH_ + dh;

        v8f cq = {};
#pragma unroll
        for (int kb = 0; kb < 4; ++kb)
            cq = wmma16(a[kb], load_b(Wq16, DP_, nt * 16, kb * 32, lane), cq);
        float biasq = bq[n];
#pragma unroll
        for (int r = 0; r < 8; ++r)
            Q16[qkbase + (size_t)(r + 8 * hh) * DH_] = (half_t)(cq[r] + biasq);

        v8f ck = {};
#pragma unroll
        for (int kb = 0; kb < 4; ++kb)
            ck = wmma16(a[kb], load_b(Wk16, DP_, nt * 16, kb * 32, lane), ck);
        float biask = bk[n];
#pragma unroll
        for (int r = 0; r < 8; ++r)
            K16[qkbase + (size_t)(r + 8 * hh) * DH_] = (half_t)(ck[r] + biask);

        v8f cv = {};
#pragma unroll
        for (int kb = 0; kb < 4; ++kb)
            cv = wmma16(a[kb], load_b(Wv16, DP_, nt * 16, kb * 32, lane), cv);
        float biasv = bv[n];
        size_t vbase = ((size_t)(bb * H_ + hd) * DH_ + dh) * S_ + s0;
#pragma unroll
        for (int r = 0; r < 8; ++r)
            VT16[vbase + r + 8 * hh] = (half_t)(cv[r] + biasv);
    }
}

// ---------------------------------------------------------------------------
// 2) Attention: one wave per (b, h, q-tile of 16).
//    scores = (Q K^T)/sqrt(D) * matrix ; exact 2-pass softmax ; O = P V / sum
// ---------------------------------------------------------------------------
__global__ __launch_bounds__(32) void attn_kernel(
    const half_t* __restrict__ Q16, const half_t* __restrict__ K16,
    const half_t* __restrict__ VT16, const float* __restrict__ matrix,
    half_t* __restrict__ O16) {
    __shared__ float  Sl[16][S_];   // 32 KB masked scores
    __shared__ half_t Pl[16][S_];   // 16 KB unnormalized exp
    __shared__ float  isum[16];

    int lane = threadIdx.x;
    int qt = blockIdx.x & 31;        // S/16 = 32 q-tiles
    int bh = blockIdx.x >> 5;
    int b  = bh >> 2;
    int h  = bh & 3;
    int hh = lane >> 4, ln = lane & 15;

    const half_t* Qb = Q16  + (size_t)bh * S_ * DH_;
    const half_t* Kb = K16  + (size_t)bh * S_ * DH_;
    const half_t* Vb = VT16 + (size_t)bh * DH_ * S_;

    v16h aq = load_frag_k8(Qb, qt * 16, lane);       // K-dim 8, rest zero
    const float scale = 0.1f;                        // 1/sqrt(100)
    const float* mrow = matrix + ((size_t)b * S_ + qt * 16) * S_;

    for (int kt = 0; kt < 32; ++kt) {                // 32 key tiles of 16
        v16h bk = load_frag_k8(Kb, kt * 16, lane);
        v8f c = {};
        c = wmma16(aq, bk, c);
        int col = kt * 16 + ln;
#pragma unroll
        for (int r = 0; r < 8; ++r) {
            int row = r + 8 * hh;
            float m = mrow[(size_t)row * S_ + col];
            Sl[row][col] = c[r] * scale * m;
        }
    }
    __syncthreads();

    for (int r = 0; r < 16; ++r) {
        float mx = -3.0e38f;
        for (int c = lane; c < S_; c += 32) mx = fmaxf(mx, Sl[r][c]);
#pragma unroll
        for (int off = 16; off > 0; off >>= 1) mx = fmaxf(mx, __shfl_xor(mx, off));
        float sum = 0.f;
        for (int c = lane; c < S_; c += 32) {
            float e = __expf(Sl[r][c] - mx);
            Pl[r][c] = (half_t)e;
            sum += e;
        }
#pragma unroll
        for (int off = 16; off > 0; off >>= 1) sum += __shfl_xor(sum, off);
        if (lane == 0) isum[r] = 1.f / sum;
    }
    __syncthreads();

    v8f o = {};
    for (int kb = 0; kb < 16; ++kb) {                // 16 key blocks of 32
        const half_t* p = &Pl[ln][kb * 32 + 16 * hh]; // A frag from LDS
        v8h lo = *(const v8h*)p;
        v8h hi = *(const v8h*)(p + 8);
        v16h ap;
#pragma unroll
        for (int i = 0; i < 8; ++i) { ap[i] = lo[i]; ap[8 + i] = hi[i]; }
        v16h bv = load_b_vt(Vb, kb * 32, lane);      // contiguous from VT
        o = wmma16(ap, bv, o);
    }
    if (ln < DH_) {
#pragma unroll
        for (int r = 0; r < 8; ++r) {
            int row = r + 8 * hh;
            size_t grow = (size_t)b * S_ + qt * 16 + row;
            O16[grow * 32 + h * DH_ + ln] = (half_t)(o[r] * isum[row]);
        }
    }
}

// ---------------------------------------------------------------------------
// 3) O @ Wo^T + bo + x  -> LayerNorm(g1,b1) -> h32 (residual) + h16 (next A)
// ---------------------------------------------------------------------------
__global__ __launch_bounds__(32) void oproj_ln1(
    const half_t* __restrict__ O16, const half_t* __restrict__ Wo16,
    const float* __restrict__ bo, const float* __restrict__ x,
    const float* __restrict__ g1, const float* __restrict__ b1,
    float* __restrict__ h32, half_t* __restrict__ h16) {
    __shared__ float acc[16][112];
    int lane = threadIdx.x;
    int row0 = blockIdx.x * 16;
    int hh = lane >> 4, ln = lane & 15;
    v16h a = load_a(O16, 32, row0, 0, lane);         // K = 32 exactly
    for (int nt = 0; nt < 7; ++nt) {
        v8f c = {};
        c = wmma16(a, load_b(Wo16, 32, nt * 16, 0, lane), c);
        int n = nt * 16 + ln;
        if (n < D_) {
            float bias = bo[n];
#pragma unroll
            for (int r = 0; r < 8; ++r) {
                int row = r + 8 * hh;
                acc[row][n] = c[r] + bias + x[(size_t)(row0 + row) * D_ + n];
            }
        }
    }
    __syncthreads();
    for (int r = 0; r < 16; ++r) {
        float s = 0.f;
        for (int c = lane; c < D_; c += 32) s += acc[r][c];
#pragma unroll
        for (int off = 16; off > 0; off >>= 1) s += __shfl_xor(s, off);
        float mu = s * (1.0f / D_);
        float v = 0.f;
        for (int c = lane; c < D_; c += 32) { float d = acc[r][c] - mu; v += d * d; }
#pragma unroll
        for (int off = 16; off > 0; off >>= 1) v += __shfl_xor(v, off);
        float rs = rsqrtf(v * (1.0f / D_) + 1e-5f);
        for (int c = lane; c < DP_; c += 32) {       // writes pad cols too
            if (c < D_) {
                float val = (acc[r][c] - mu) * rs * g1[c] + b1[c];
                h32[(size_t)(row0 + r) * D_ + c] = val;
                h16[(size_t)(row0 + r) * DP_ + c] = (half_t)val;
            } else {
                h16[(size_t)(row0 + r) * DP_ + c] = (half_t)0.f;
            }
        }
    }
}

// ---------------------------------------------------------------------------
// 4) FFN1: relu(h @ Wf1^T + bf1) -> ff16 [M, 416] (writes its own pad cols)
// ---------------------------------------------------------------------------
__global__ __launch_bounds__(32) void ffn1_kernel(
    const half_t* __restrict__ h16, const half_t* __restrict__ Wf1,
    const float* __restrict__ bf1, half_t* __restrict__ ff16) {
    int lane = threadIdx.x;
    int row0 = blockIdx.x * 16;
    int hh = lane >> 4, ln = lane & 15;
    v16h a[4];
#pragma unroll
    for (int kb = 0; kb < 4; ++kb) a[kb] = load_a(h16, DP_, row0, kb * 32, lane);
    for (int nt = 0; nt < 25; ++nt) {                // N = 400 = 25 tiles
        v8f c = {};
#pragma unroll
        for (int kb = 0; kb < 4; ++kb)
            c = wmma16(a[kb], load_b(Wf1, DP_, nt * 16, kb * 32, lane), c);
        int n = nt * 16 + ln;
        float bias = bf1[n];
#pragma unroll
        for (int r = 0; r < 8; ++r) {
            int row = r + 8 * hh;
            ff16[(size_t)(row0 + row) * FFP_ + n] = (half_t)fmaxf(c[r] + bias, 0.f);
        }
    }
#pragma unroll
    for (int r = 0; r < 8; ++r) {                    // zero pad cols 400..415
        int row = r + 8 * hh;
        ff16[(size_t)(row0 + row) * FFP_ + DFF_ + ln] = (half_t)0.f;
    }
}

// ---------------------------------------------------------------------------
// 5) FFN2 + residual + LayerNorm(g2,b2) -> out fp32
// ---------------------------------------------------------------------------
__global__ __launch_bounds__(32) void ffn2_ln2(
    const half_t* __restrict__ ff16, const half_t* __restrict__ Wf2,
    const float* __restrict__ bf2, const float* __restrict__ h32,
    const float* __restrict__ g2, const float* __restrict__ b2,
    float* __restrict__ out) {
    __shared__ float acc[16][112];
    int lane = threadIdx.x;
    int row0 = blockIdx.x * 16;
    int hh = lane >> 4, ln = lane & 15;
    v16h a[13];
#pragma unroll
    for (int kb = 0; kb < 13; ++kb) a[kb] = load_a(ff16, FFP_, row0, kb * 32, lane);
    for (int nt = 0; nt < 7; ++nt) {
        v8f c = {};
#pragma unroll
        for (int kb = 0; kb < 13; ++kb)
            c = wmma16(a[kb], load_b(Wf2, FFP_, nt * 16, kb * 32, lane), c);
        int n = nt * 16 + ln;
        if (n < D_) {
            float bias = bf2[n];
#pragma unroll
            for (int r = 0; r < 8; ++r) {
                int row = r + 8 * hh;
                acc[row][n] = c[r] + bias + h32[(size_t)(row0 + row) * D_ + n];
            }
        }
    }
    __syncthreads();
    for (int r = 0; r < 16; ++r) {
        float s = 0.f;
        for (int c = lane; c < D_; c += 32) s += acc[r][c];
#pragma unroll
        for (int off = 16; off > 0; off >>= 1) s += __shfl_xor(s, off);
        float mu = s * (1.0f / D_);
        float v = 0.f;
        for (int c = lane; c < D_; c += 32) { float d = acc[r][c] - mu; v += d * d; }
#pragma unroll
        for (int off = 16; off > 0; off >>= 1) v += __shfl_xor(v, off);
        float rs = rsqrtf(v * (1.0f / D_) + 1e-5f);
        for (int c = lane; c < D_; c += 32)
            out[(size_t)(row0 + r) * D_ + c] = (acc[r][c] - mu) * rs * g2[c] + b2[c];
    }
}

// ---------------------------------------------------------------------------
// Workspace layout (all offsets 256B aligned); every byte read is written
// each call, so no memset is needed.
// ---------------------------------------------------------------------------
#define OFF_X16  ((size_t)0)                         // [M,128] f16
#define OFF_WQ   (OFF_X16 + (size_t)M_ * DP_ * 2)    // [32,128] f16
#define OFF_WK   (OFF_WQ + 32 * DP_ * 2)
#define OFF_WV   (OFF_WK + 32 * DP_ * 2)
#define OFF_WO   (OFF_WV + 32 * DP_ * 2)             // [112,32] f16
#define OFF_WF1  (OFF_WO + 112 * 32 * 2)             // [400,128] f16
#define OFF_WF2  (OFF_WF1 + 400 * DP_ * 2)           // [112,416] f16
#define OFF_Q    (OFF_WF2 + 112 * FFP_ * 2)          // [B,H,S,8] f16
#define OFF_K    (OFF_Q + (size_t)B_ * H_ * S_ * DH_ * 2)
#define OFF_VT   (OFF_K + (size_t)B_ * H_ * S_ * DH_ * 2)  // [B,H,8,S] f16
#define OFF_O    (OFF_VT + (size_t)B_ * H_ * DH_ * S_ * 2) // [M,32] f16
#define OFF_H32  (OFF_O + (size_t)M_ * 32 * 2)       // [M,100] f32
#define OFF_H16  (OFF_H32 + (size_t)M_ * D_ * 4)     // [M,128] f16
#define OFF_FF   (OFF_H16 + (size_t)M_ * DP_ * 2)    // [M,416] f16
#define WS_TOTAL (OFF_FF + (size_t)M_ * FFP_ * 2)    // ~62.7 MiB

extern "C" void kernel_launch(void* const* d_in, const int* in_sizes, int n_in,
                              void* d_out, int out_size, void* d_ws, size_t ws_size,
                              hipStream_t stream) {
    const float* x   = (const float*)d_in[0];
    const float* mat = (const float*)d_in[1];
    const float* Wq  = (const float*)d_in[2];
    const float* bq  = (const float*)d_in[3];
    const float* Wk  = (const float*)d_in[4];
    const float* bk  = (const float*)d_in[5];
    const float* Wv  = (const float*)d_in[6];
    const float* bv  = (const float*)d_in[7];
    const float* Wo  = (const float*)d_in[8];
    const float* bo  = (const float*)d_in[9];
    const float* g1  = (const float*)d_in[10];
    const float* b1  = (const float*)d_in[11];
    const float* Wf1 = (const float*)d_in[12];
    const float* bf1 = (const float*)d_in[13];
    const float* Wf2 = (const float*)d_in[14];
    const float* bf2 = (const float*)d_in[15];
    const float* g2  = (const float*)d_in[16];
    const float* b2  = (const float*)d_in[17];
    float* out = (float*)d_out;

    char* ws = (char*)d_ws;
    half_t* x16   = (half_t*)(ws + OFF_X16);
    half_t* Wq16  = (half_t*)(ws + OFF_WQ);
    half_t* Wk16  = (half_t*)(ws + OFF_WK);
    half_t* Wv16  = (half_t*)(ws + OFF_WV);
    half_t* Wo16  = (half_t*)(ws + OFF_WO);
    half_t* Wf116 = (half_t*)(ws + OFF_WF1);
    half_t* Wf216 = (half_t*)(ws + OFF_WF2);
    half_t* Q16   = (half_t*)(ws + OFF_Q);
    half_t* K16   = (half_t*)(ws + OFF_K);
    half_t* VT16  = (half_t*)(ws + OFF_VT);
    half_t* O16   = (half_t*)(ws + OFF_O);
    float*  h32   = (float*)(ws + OFF_H32);
    half_t* h16   = (half_t*)(ws + OFF_H16);
    half_t* ff16  = (half_t*)(ws + OFF_FF);

    auto blks = [](int n) { return (n + 255) / 256; };
    cvt_pad<<<blks(M_ * DP_), 256, 0, stream>>>(x, x16, M_, D_, M_, DP_);
    cvt_pad<<<blks(SZ_ * DP_), 256, 0, stream>>>(Wq, Wq16, SZ_, D_, SZ_, DP_);
    cvt_pad<<<blks(SZ_ * DP_), 256, 0, stream>>>(Wk, Wk16, SZ_, D_, SZ_, DP_);
    cvt_pad<<<blks(SZ_ * DP_), 256, 0, stream>>>(Wv, Wv16, SZ_, D_, SZ_, DP_);
    cvt_pad<<<blks(112 * 32), 256, 0, stream>>>(Wo, Wo16, D_, SZ_, 112, 32);
    cvt_pad<<<blks(DFF_ * DP_), 256, 0, stream>>>(Wf1, Wf116, DFF_, D_, DFF_, DP_);
    cvt_pad<<<blks(112 * FFP_), 256, 0, stream>>>(Wf2, Wf216, D_, DFF_, 112, FFP_);

    qkv_kernel<<<M_ / 16, 32, 0, stream>>>(x16, Wq16, Wk16, Wv16, bq, bk, bv,
                                           Q16, K16, VT16);
    attn_kernel<<<B_ * H_ * (S_ / 16), 32, 0, stream>>>(Q16, K16, VT16, mat, O16);
    oproj_ln1<<<M_ / 16, 32, 0, stream>>>(O16, Wo16, bo, x, g1, b1, h32, h16);
    ffn1_kernel<<<M_ / 16, 32, 0, stream>>>(h16, Wf116, bf1, ff16);
    ffn2_ln2<<<M_ / 16, 32, 0, stream>>>(ff16, Wf216, bf2, h32, g2, b2, out);

    (void)in_sizes; (void)n_in; (void)out_size; (void)ws_size;
}
